// MultiHeadAttention_28226525069348
// MI455X (gfx1250) — matrix operand order repaired and checked by reading, not compiled, not run on hardware
//
#include <hip/hip_runtime.h>
#include <hip/hip_bf16.h>

// ---------------------------------------------------------------------------
// MI455X (gfx1250) fused multi-head attention, bf16 WMMA (16x16x32, f32 acc).
//
//   K1: qkv_proj   X[8192,1024]f32 @ W^T + b -> Q,K bf16 [B,H,S,Hd]; V bf16 [B,H,Hd,S]
//   K2: attn       flash-attention per (b,h,128 q rows), online softmax,
//                  double-buffered GLOBAL_LOAD_ASYNC_TO_LDS staging of K/V tiles
//   K3: out_proj   ctx[8192,1024]bf16 @ Wo^T + bo -> f32 out [B,S,D]
//
// Workspace usage (d_ws): 4 buffers of 8192*1024 bf16 = 64 MB total.
// ---------------------------------------------------------------------------

typedef __bf16 v16bf __attribute__((ext_vector_type(16)));
typedef float  v8f   __attribute__((ext_vector_type(8)));
typedef int    v4i   __attribute__((ext_vector_type(4)));

union Frag {            // one WMMA A/B operand: 16 bf16 = 32 bytes
    v16bf v;
    uint4 q[2];
};

__device__ __forceinline__ __bf16 f2bf(float f) { return (__bf16)f; }

__device__ __forceinline__ v8f wmma_bf16(const Frag& a, const Frag& b, v8f c) {
    // D = A(16x32 bf16) * B(32x16 bf16) + C(16x16 f32)
    return __builtin_amdgcn_wmma_f32_16x16x32_bf16(
        /*neg_a=*/false, a.v, /*neg_b=*/false, b.v,
        /*c_mod=*/(short)0, c, /*reuse_a=*/false, /*reuse_b=*/false);
}

// --- CDNA5 async global->LDS copy (ASYNCcnt tracked), with safe fallback ---
#if __has_builtin(__builtin_amdgcn_global_load_async_to_lds_b128)
#define ASYNC_CP_B128(dst_lds, src_glob)                                       \
    __builtin_amdgcn_global_load_async_to_lds_b128(                            \
        (__attribute__((address_space(1))) v4i*)(unsigned long long)(src_glob),\
        (__attribute__((address_space(3))) v4i*)(unsigned long long)(dst_lds), \
        0, 0)
#else
#define ASYNC_CP_B128(dst_lds, src_glob)                                       \
    (*(uint4*)(dst_lds) = *(const uint4*)(src_glob))
#endif

__device__ __forceinline__ void wait_async0() {
    asm volatile("s_wait_asynccnt 0x0" ::: "memory");
}

// Problem constants
#define BATCH 4
#define SEQ   2048
#define DMODEL 1024
#define NHEAD 16
#define HDIM  64
#define MROWS (BATCH * SEQ)     // 8192

// ===========================================================================
// Kernel 1: QKV projection.  Y[m][n] = sum_k X[m][k]*W[n][k] + b[n]
// blockDim 256 (8 waves). Workgroup tile 128x128; wave tile 32x64.
// blockIdx.z selects (query,Wq,bq)->Q, (key,Wk,bk)->K, (value,Wv,bv)->V.
// Q,K written [B,H,S,Hd]; V written transposed [B,H,Hd,S] so the attention
// kernel can stage V tiles with straight (async) b128 copies.
// Global->LDS staging is software-pipelined through registers.
// ===========================================================================
__global__ __launch_bounds__(256) void qkv_proj(
    const float* __restrict__ Xq, const float* __restrict__ Xk,
    const float* __restrict__ Xv,
    const float* __restrict__ Wq, const float* __restrict__ bq,
    const float* __restrict__ Wk, const float* __restrict__ bk,
    const float* __restrict__ Wv, const float* __restrict__ bv,
    __bf16* __restrict__ qb, __bf16* __restrict__ kb, __bf16* __restrict__ vb)
{
    const int z = blockIdx.z;
    const float* X    = (z == 0) ? Xq : (z == 1) ? Xk : Xv;
    const float* W    = (z == 0) ? Wq : (z == 1) ? Wk : Wv;
    const float* bias = (z == 0) ? bq : (z == 1) ? bk : bv;
    __bf16* out       = (z == 0) ? qb : (z == 1) ? kb : vb;

    __shared__ __align__(16) __bf16 Xs[128 * 32];   // [row][k] 8 KB
    __shared__ __align__(16) __bf16 Ws[128 * 32];   // [n][k]   8 KB

    const int tid  = threadIdx.x;
    const int lane = tid & 31, w = tid >> 5;
    const int half = lane >> 4, l16 = lane & 15;
    const int wrow = w >> 1, wcol = w & 1;          // 4x2 wave grid
    const int m0 = blockIdx.y * 128;
    const int n0 = blockIdx.x * 128;
    const int mb = m0 + wrow * 32;
    const int nb = n0 + wcol * 64;

    v8f acc[2][4];
    for (int mt = 0; mt < 2; ++mt)
        for (int nt = 0; nt < 4; ++nt)
            for (int r = 0; r < 8; ++r) acc[mt][nt][r] = 0.0f;

    // staging: each thread owns half a 32-float row (16 floats) per tile
    const int srow = tid >> 1;
    const int scol = (tid & 1) * 16;
    const float* xg0 = X + (size_t)(m0 + srow) * DMODEL + scol;
    const float* wg0 = W + (size_t)(n0 + srow) * DMODEL + scol;

    float4 xr[4], wr[4];
    for (int j = 0; j < 4; ++j) {                   // prologue: tile kk=0
        xr[j] = ((const float4*)xg0)[j];
        wr[j] = ((const float4*)wg0)[j];
    }

    for (int kk = 0; kk < DMODEL; kk += 32) {
        __syncthreads();                            // prev iter frag reads done
        for (int j = 0; j < 4; ++j) {               // regs -> LDS (f32->bf16)
            union { __bf16 h[4]; uint2 u; } px, pw;
            px.h[0] = f2bf(xr[j].x); px.h[1] = f2bf(xr[j].y);
            px.h[2] = f2bf(xr[j].z); px.h[3] = f2bf(xr[j].w);
            pw.h[0] = f2bf(wr[j].x); pw.h[1] = f2bf(wr[j].y);
            pw.h[2] = f2bf(wr[j].z); pw.h[3] = f2bf(wr[j].w);
            *(uint2*)&Xs[srow * 32 + scol + j * 4] = px.u;
            *(uint2*)&Ws[srow * 32 + scol + j * 4] = pw.u;
        }
        __syncthreads();

        if (kk + 32 < DMODEL) {                     // prefetch next tile
            for (int j = 0; j < 4; ++j) {
                xr[j] = ((const float4*)(xg0 + kk + 32))[j];
                wr[j] = ((const float4*)(wg0 + kk + 32))[j];
            }
        }

        Frag aF[2], bF[4];
        for (int mt = 0; mt < 2; ++mt) {
            const __bf16* base = &Xs[(wrow * 32 + mt * 16 + l16) * 32];
            aF[mt].q[0] = *(const uint4*)(base + half * 8);        // K = h*8+0..7
            aF[mt].q[1] = *(const uint4*)(base + 16 + half * 8);   // K = 16+h*8+..
        }
        for (int nt = 0; nt < 4; ++nt) {
            const __bf16* base = &Ws[(wcol * 64 + nt * 16 + l16) * 32 + half * 8];
            bF[nt].q[0] = *(const uint4*)(base);                   // K = h*8+0..7
            bF[nt].q[1] = *(const uint4*)(base + 16);              // K = 16+h*8+..
        }
        for (int mt = 0; mt < 2; ++mt)
            for (int nt = 0; nt < 4; ++nt)
                acc[mt][nt] = wmma_bf16(aF[mt], bF[nt], acc[mt][nt]);
    }

    // epilogue: add bias, scatter bf16 (Q/K: [B,H,S,Hd]; V: [B,H,Hd,S])
    for (int nt = 0; nt < 4; ++nt) {
        const int n = nb + nt * 16 + l16;           // 0..1023
        const float bv_ = bias[n];
        const int h = n >> 6, hd = n & 63;
        for (int mt = 0; mt < 2; ++mt) {
            for (int r = 0; r < 8; ++r) {
                const int m = mb + mt * 16 + r + half * 8;
                const int b = m >> 11, s = m & (SEQ - 1);
                const size_t idx = (z == 2)
                    ? (((size_t)b * NHEAD + h) * HDIM + hd) * SEQ + s
                    : (((size_t)b * NHEAD + h) * SEQ + s) * HDIM + hd;
                out[idx] = f2bf(acc[mt][nt][r] + bv_);
            }
        }
    }
}

// ===========================================================================
// Kernel 2: fused flash attention. grid = (S/128, H, B), block = 256 (8 waves)
// Each wave owns 16 q rows; all 8 waves share the staged 32-key K/V tiles.
// K/V tiles staged with async global->LDS copies, double buffered so the
// copy of tile kt+1 runs under the 8 WMMAs of tile kt.
// ===========================================================================
__global__ __launch_bounds__(256) void attn(
    const __bf16* __restrict__ qb, const __bf16* __restrict__ kb,
    const __bf16* __restrict__ vb, const int* __restrict__ mask,
    __bf16* __restrict__ cb)
{
    const int h = blockIdx.y, b = blockIdx.z;
    const int tid  = threadIdx.x;
    const int lane = tid & 31, w = tid >> 5;
    const int half = lane >> 4, l16 = lane & 15;

    const size_t bh = (size_t)b * NHEAD + h;
    const __bf16* Qh  = qb + bh * SEQ * HDIM;       // [S][64]
    const __bf16* Kh  = kb + bh * SEQ * HDIM;       // [S][64]
    const __bf16* Vth = vb + bh * HDIM * SEQ;       // [64][S] (pre-transposed)
    const int*    mk  = mask + (size_t)b * SEQ;

    __shared__ __align__(16) __bf16 Ks[2][32 * 64]; // [key][hd]   2 x 4 KB
    __shared__ __align__(16) __bf16 Vt[2][64 * 32]; // [hd][key]   2 x 4 KB
    __shared__ __align__(16) __bf16 Ps[8][16 * 32]; // per-wave P  8 KB

    const int q0 = blockIdx.x * 128 + w * 16;

    // Q fragments (held for the whole key loop): A = Q[16 x 64] in 2 k-slices
    Frag aQ0, aQ1;
    {
        const __bf16* qrow = Qh + (size_t)(q0 + l16) * HDIM;
        aQ0.q[0] = *(const uint4*)(qrow + half * 8);
        aQ0.q[1] = *(const uint4*)(qrow + 16 + half * 8);
        aQ1.q[0] = *(const uint4*)(qrow + 32 + half * 8);
        aQ1.q[1] = *(const uint4*)(qrow + 48 + half * 8);
    }

    float Mr[8], Lr[8];
    v8f o[4];
    for (int r = 0; r < 8; ++r) { Mr[r] = -3.0e38f; Lr[r] = 0.0f; }
    for (int nt = 0; nt < 4; ++nt)
        for (int r = 0; r < 8; ++r) o[nt][r] = 0.0f;

    // staging split: K tile 32x64 (16B/thread), V tile 64x32 (16B/thread)
    const int ksr = tid >> 3, ksc = (tid & 7) * 8;  // K: row 0..31, col 0..56
    const int vsr = tid >> 2, vsc = (tid & 3) * 8;  // V: row 0..63, col 0..24

    // prologue: issue async copies for tile 0 into buffer 0
    ASYNC_CP_B128(&Ks[0][ksr * 64 + ksc], Kh + (size_t)ksr * HDIM + ksc);
    ASYNC_CP_B128(&Vt[0][vsr * 32 + vsc], Vth + (size_t)vsr * SEQ + vsc);

    for (int kt = 0; kt < SEQ / 32; ++kt) {
        const int cur = kt & 1;
        wait_async0();                              // tile kt resident
        __syncthreads();                            // visible to all 8 waves

        if (kt + 1 < SEQ / 32) {                    // overlap copy of kt+1
            const int nxt = cur ^ 1;
            ASYNC_CP_B128(&Ks[nxt][ksr * 64 + ksc],
                          Kh + (size_t)((kt + 1) * 32 + ksr) * HDIM + ksc);
            ASYNC_CP_B128(&Vt[nxt][vsr * 32 + vsc],
                          Vth + (size_t)vsr * SEQ + (kt + 1) * 32 + vsc);
        }

        // ---- scores S = Q * K^T  (16 x 32, two 16x16 accs) --------------
        v8f s0, s1;
        for (int r = 0; r < 8; ++r) { s0[r] = 0.0f; s1[r] = 0.0f; }
        {
            const __bf16* kr = &Ks[cur][l16 * 64];          // keys 0..15
            Frag bk0, bk1;
            bk0.q[0] = *(const uint4*)(kr + half * 8);
            bk0.q[1] = *(const uint4*)(kr + 16 + half * 8);
            bk1.q[0] = *(const uint4*)(kr + 32 + half * 8);
            bk1.q[1] = *(const uint4*)(kr + 48 + half * 8);
            s0 = wmma_bf16(aQ0, bk0, s0);
            s0 = wmma_bf16(aQ1, bk1, s0);
        }
        {
            const __bf16* kr = &Ks[cur][(16 + l16) * 64];   // keys 16..31
            Frag bk0, bk1;
            bk0.q[0] = *(const uint4*)(kr + half * 8);
            bk0.q[1] = *(const uint4*)(kr + 16 + half * 8);
            bk1.q[0] = *(const uint4*)(kr + 32 + half * 8);
            bk1.q[1] = *(const uint4*)(kr + 48 + half * 8);
            s1 = wmma_bf16(aQ0, bk0, s1);
            s1 = wmma_bf16(aQ1, bk1, s1);
        }

        // ---- scale + mask (mask indexed by key column) ------------------
        const int msk0 = mk[kt * 32 + l16];
        const int msk1 = mk[kt * 32 + 16 + l16];
        for (int r = 0; r < 8; ++r) {
            s0[r] = msk0 ? s0[r] * 0.125f : -1.0e10f;   // 1/sqrt(64)
            s1[r] = msk1 ? s1[r] * 0.125f : -1.0e10f;
        }

        // ---- online softmax (row stats replicated over 16-lane halves) --
        for (int r = 0; r < 8; ++r) {
            float t = fmaxf(s0[r], s1[r]);
            t = fmaxf(t, __shfl_xor(t, 1, 32));
            t = fmaxf(t, __shfl_xor(t, 2, 32));
            t = fmaxf(t, __shfl_xor(t, 4, 32));
            t = fmaxf(t, __shfl_xor(t, 8, 32));
            const float nm   = fmaxf(Mr[r], t);
            const float corr = __expf(Mr[r] - nm);
            const float p0 = __expf(s0[r] - nm);
            const float p1 = __expf(s1[r] - nm);
            float ps = p0 + p1;
            ps += __shfl_xor(ps, 1, 32);
            ps += __shfl_xor(ps, 2, 32);
            ps += __shfl_xor(ps, 4, 32);
            ps += __shfl_xor(ps, 8, 32);
            Lr[r] = Lr[r] * corr + ps;
            Mr[r] = nm;
            for (int nt = 0; nt < 4; ++nt) o[nt][r] *= corr;
            // C-frag -> LDS (row m, key col) for A-frag relayout
            const int m = r + half * 8;
            Ps[w][m * 32 + l16]      = f2bf(p0);
            Ps[w][m * 32 + 16 + l16] = f2bf(p1);
        }

        // ---- O += P * V  (A = P 16x32, B = V 32x16 per hd-tile) ---------
        Frag aP;
        {
            const __bf16* pr = &Ps[w][l16 * 32];
            aP.q[0] = *(const uint4*)(pr + half * 8);
            aP.q[1] = *(const uint4*)(pr + 16 + half * 8);
        }
        for (int nt = 0; nt < 4; ++nt) {
            const __bf16* vr = &Vt[cur][(nt * 16 + l16) * 32 + half * 8];
            Frag bV;
            bV.q[0] = *(const uint4*)(vr);
            bV.q[1] = *(const uint4*)(vr + 16);
            o[nt] = wmma_bf16(aP, bV, o[nt]);
        }
    }

    // ---- normalize and store context as bf16 [B,S,D] ----------------------
    for (int r = 0; r < 8; ++r) {
        const float inv = 1.0f / Lr[r];
        const int qrow = q0 + r + half * 8;
        const size_t base = ((size_t)b * SEQ + qrow) * DMODEL + h * HDIM;
        for (int nt = 0; nt < 4; ++nt)
            cb[base + nt * 16 + l16] = f2bf(o[nt][r] * inv);
    }
}

// ===========================================================================
// Kernel 3: output projection. ctx bf16 [8192,1024] @ Wo^T + bo -> f32 out.
// A-fragments loaded directly from global (bf16, layout-compatible);
// Wo converted fp32->bf16 through LDS with register-pipelined staging.
// ===========================================================================
__global__ __launch_bounds__(256) void out_proj(
    const __bf16* __restrict__ ctx, const float* __restrict__ Wo,
    const float* __restrict__ bo, float* __restrict__ out)
{
    __shared__ __align__(16) __bf16 Ws[128 * 32];

    const int tid  = threadIdx.x;
    const int lane = tid & 31, w = tid >> 5;
    const int half = lane >> 4, l16 = lane & 15;
    const int wrow = w >> 1, wcol = w & 1;
    const int m0 = blockIdx.y * 128;
    const int n0 = blockIdx.x * 128;
    const int mb = m0 + wrow * 32;
    const int nb = n0 + wcol * 64;

    v8f acc[2][4];
    for (int mt = 0; mt < 2; ++mt)
        for (int nt = 0; nt < 4; ++nt)
            for (int r = 0; r < 8; ++r) acc[mt][nt][r] = 0.0f;

    const int srow = tid >> 1;
    const int scol = (tid & 1) * 16;
    const float* wg0 = Wo + (size_t)(n0 + srow) * DMODEL + scol;

    float4 wr[4];
    for (int j = 0; j < 4; ++j) wr[j] = ((const float4*)wg0)[j];

    for (int kk = 0; kk < DMODEL; kk += 32) {
        __syncthreads();
        for (int j = 0; j < 4; ++j) {
            union { __bf16 h[4]; uint2 u; } pw;
            pw.h[0] = f2bf(wr[j].x); pw.h[1] = f2bf(wr[j].y);
            pw.h[2] = f2bf(wr[j].z); pw.h[3] = f2bf(wr[j].w);
            *(uint2*)&Ws[srow * 32 + scol + j * 4] = pw.u;
        }
        __syncthreads();

        if (kk + 32 < DMODEL)
            for (int j = 0; j < 4; ++j)
                wr[j] = ((const float4*)(wg0 + kk + 32))[j];

        Frag aF[2], bF[4];
        for (int mt = 0; mt < 2; ++mt) {
            const __bf16* base = ctx + (size_t)(mb + mt * 16 + l16) * DMODEL + kk;
            aF[mt].q[0] = *(const uint4*)(base + half * 8);
            aF[mt].q[1] = *(const uint4*)(base + 16 + half * 8);
        }
        for (int nt = 0; nt < 4; ++nt) {
            const __bf16* base = &Ws[(wcol * 64 + nt * 16 + l16) * 32 + half * 8];
            bF[nt].q[0] = *(const uint4*)(base);
            bF[nt].q[1] = *(const uint4*)(base + 16);
        }
        for (int mt = 0; mt < 2; ++mt)
            for (int nt = 0; nt < 4; ++nt)
                acc[mt][nt] = wmma_bf16(aF[mt], bF[nt], acc[mt][nt]);
    }

    for (int nt = 0; nt < 4; ++nt) {
        const int n = nb + nt * 16 + l16;
        const float bv_ = bo[n];
        for (int mt = 0; mt < 2; ++mt)
            for (int r = 0; r < 8; ++r) {
                const int m = mb + mt * 16 + r + half * 8;
                out[(size_t)m * DMODEL + n] = acc[mt][nt][r] + bv_;
            }
    }
}

// ===========================================================================
// Launch
// ===========================================================================
extern "C" void kernel_launch(void* const* d_in, const int* in_sizes, int n_in,
                              void* d_out, int out_size, void* d_ws, size_t ws_size,
                              hipStream_t stream)
{
    const float* query = (const float*)d_in[0];
    const float* key   = (const float*)d_in[1];
    const float* value = (const float*)d_in[2];
    const int*   mask  = (const int*)d_in[3];
    const float* Wq    = (const float*)d_in[4];
    const float* bq    = (const float*)d_in[5];
    const float* Wk    = (const float*)d_in[6];
    const float* bk    = (const float*)d_in[7];
    const float* Wv    = (const float*)d_in[8];
    const float* bv    = (const float*)d_in[9];
    const float* Wo    = (const float*)d_in[10];
    const float* bo    = (const float*)d_in[11];
    float* out = (float*)d_out;

    // workspace: Q,K (bf16 [B,H,S,Hd]), V (bf16 [B,H,Hd,S]), ctx (bf16 [B,S,D])
    const size_t elems = (size_t)MROWS * DMODEL;
    __bf16* qbuf = (__bf16*)d_ws;
    __bf16* kbuf = qbuf + elems;
    __bf16* vbuf = kbuf + elems;
    __bf16* cbuf = vbuf + elems;

    dim3 blk(256);

    dim3 g1(DMODEL / 128, MROWS / 128, 3);           // (8, 64, 3)
    qkv_proj<<<g1, blk, 0, stream>>>(query, key, value,
                                     Wq, bq, Wk, bk, Wv, bv,
                                     qbuf, kbuf, vbuf);

    dim3 g2(SEQ / 128, NHEAD, BATCH);                // (16, 16, 4)
    attn<<<g2, blk, 0, stream>>>(qbuf, kbuf, vbuf, mask, cbuf);

    dim3 g3(DMODEL / 128, MROWS / 128);              // (8, 64)
    out_proj<<<g3, blk, 0, stream>>>(cbuf, Wo, bo, out);
}
